// Block_42348377539010
// MI455X (gfx1250) — matrix operand order repaired and checked by reading, not compiled, not run on hardware
//
#include <hip/hip_runtime.h>
#include <hip/hip_bf16.h>
#include <math.h>
#include <stdint.h>

typedef __attribute__((ext_vector_type(16))) _Float16 v16h;
typedef __attribute__((ext_vector_type(8)))  _Float16 v8h;
typedef __attribute__((ext_vector_type(8)))  float    v8f;
typedef __attribute__((ext_vector_type(4)))  int      v4i;

#define WMMA_F16(a, b, c) \
  __builtin_amdgcn_wmma_f32_16x16x32_f16(false, (a), false, (b), (short)0, (c), false, false)

// ---------------------------------------------------------------------------
// Async Global -> LDS copy (16B per lane), CDNA5 GLOBAL_LOAD_ASYNC_TO_LDS_B128.
// Builtin takes typed b128 (v4i) pointers: global AS(1) src, LDS AS(3) dst.
// Falls back to a synchronous copy if the builtin is absent on this toolchain.
// ---------------------------------------------------------------------------
#if __has_builtin(__builtin_amdgcn_global_load_async_to_lds_b128)
#define HAS_ASYNC 1
#endif

__device__ __forceinline__ void copy16_g2l(const _Float16* g, _Float16* l) {
#ifdef HAS_ASYNC
  // Flat LDS addresses carry the LDS byte offset in the low 32 bits (ISA 10.2);
  // global flat addresses are identity-mapped.
  auto gp = (__attribute__((address_space(1))) v4i*)(unsigned long long)(uintptr_t)g;
  auto lp = (__attribute__((address_space(3))) v4i*)(unsigned int)(uintptr_t)l;
  __builtin_amdgcn_global_load_async_to_lds_b128(gp, lp, 0, 0);
#else
  *(v8h*)l = *(const v8h*)g;
#endif
}

#ifdef HAS_ASYNC
#if __has_builtin(__builtin_amdgcn_s_wait_asynccnt)
#define WAIT_ASYNC(imm) __builtin_amdgcn_s_wait_asynccnt(imm)
#else
#define WAIT_ASYNC(imm) asm volatile("s_wait_asynccnt " #imm ::: "memory")
#endif
#else
#define WAIT_ASYNC(imm)
#endif

// Build a 16-half A/B fragment from two contiguous 8-half (16B) chunks.
__device__ __forceinline__ v16h frag_from2(const _Float16* p0, const _Float16* p1) {
  union { v8h h[2]; v16h f; } u;
  u.h[0] = *(const v8h*)p0;
  u.h[1] = *(const v8h*)p1;
  return u.f;
}

// ---------------------------------------------------------------------------
// Weight cast + transpose: W[K,N] f32  ->  WT[N,K] f16   (32x32 LDS tiles)
// ---------------------------------------------------------------------------
__global__ __launch_bounds__(256)
void k_transpose_cast(const float* __restrict__ W, _Float16* __restrict__ WT,
                      int K, int N) {
  __shared__ _Float16 tile[32][33];
  const int kb = blockIdx.y * 32, nb = blockIdx.x * 32;
  const int tx = threadIdx.x & 31, ty = threadIdx.x >> 5;
  for (int i = ty; i < 32; i += 8)
    tile[i][tx] = (_Float16)W[(size_t)(kb + i) * N + nb + tx];
  __syncthreads();
  for (int i = ty; i < 32; i += 8)
    WT[(size_t)(nb + i) * K + kb + tx] = tile[tx][i];
}

// ---------------------------------------------------------------------------
// LayerNorm (row of C floats) + cast to f16.  One 256-thread block per row.
// ---------------------------------------------------------------------------
__global__ __launch_bounds__(256)
void k_ln_cast(const float* __restrict__ X, const float* __restrict__ w,
               const float* __restrict__ bb, _Float16* __restrict__ outp, int C) {
  __shared__ float red[16];
  const int row = blockIdx.x;
  const float* xr = X + (size_t)row * C;
  float s = 0.f, s2 = 0.f;
  for (int i = threadIdx.x; i < C; i += blockDim.x) {
    float v = xr[i]; s += v; s2 += v * v;
  }
  for (int o = 16; o > 0; o >>= 1) { s += __shfl_down(s, o); s2 += __shfl_down(s2, o); }
  const int lane = threadIdx.x & 31, wv = threadIdx.x >> 5;
  if (lane == 0) { red[wv] = s; red[8 + wv] = s2; }
  __syncthreads();
  if (threadIdx.x == 0) {
    float ts = 0.f, t2 = 0.f;
    for (int i = 0; i < 8; i++) { ts += red[i]; t2 += red[8 + i]; }
    red[0] = ts; red[8] = t2;
  }
  __syncthreads();
  const float mu  = red[0] / C;
  const float var = red[8] / C - mu * mu;
  const float inv = rsqrtf(var + 1e-5f);
  _Float16* orow = outp + (size_t)row * C;
  for (int i = threadIdx.x; i < C; i += blockDim.x)
    orow[i] = (_Float16)((xr[i] - mu) * inv * w[i] + bb[i]);
}

// ---------------------------------------------------------------------------
// WMMA GEMM:  C[M,N](fp32 acc) = A16[M,K] @ BT16[N,K]^T + bias, with epilogue.
// Block = 128 threads = 4 waves (2x2); block tile 128x128, wave tile 64x64.
// K staged in 32-wide slices through LDS, double buffered, async-copied with
// GLOBAL_LOAD_ASYNC_TO_LDS_B128 + s_wait_asynccnt (ASYNCcnt in-order).
// ---------------------------------------------------------------------------
enum { EPI_H16 = 0, EPI_RES = 1, EPI_GELU = 2 };

template <int EPI>
__global__ __launch_bounds__(128)
void k_gemm(const _Float16* __restrict__ A, const _Float16* __restrict__ BT,
            const float* __restrict__ bias, const float* __restrict__ resid,
            float* __restrict__ Cf, _Float16* __restrict__ Ch,
            int M, int N, int K) {
  // 40-half row stride: 16B-aligned chunks, conflict-free 16-lane access.
  __shared__ __align__(16) _Float16 sA[2][128][40];
  __shared__ __align__(16) _Float16 sB[2][128][40];

  const int tid  = threadIdx.x;
  const int lane = tid & 31;
  const int wave = tid >> 5;
  const int r = lane & 15, half = lane >> 4;
  const int mBlock = blockIdx.y * 128;
  const int nBlock = blockIdx.x * 128;
  const int mL = (wave >> 1) * 64;
  const int nL = (wave & 1) * 64;
  const int NS = K >> 5;  // 32-wide K stages

  // Stage copier: 128 threads move A[128x32] + B[128x32] f16 slices.
  // Per thread: 4 A-chunks + 4 B-chunks of 16B -> 8 async instrs per wave/stage.
  auto issue = [&](int s, int buf) {
#pragma unroll
    for (int c = 0; c < 4; c++) {
      const int chunk = c * 128 + tid;        // 0..511
      const int row = chunk >> 2, kc = chunk & 3;
      copy16_g2l(A  + (size_t)(mBlock + row) * K + s * 32 + kc * 8, &sA[buf][row][kc * 8]);
      copy16_g2l(BT + (size_t)(nBlock + row) * K + s * 32 + kc * 8, &sB[buf][row][kc * 8]);
    }
  };

  v8f acc[4][4] = {};
  issue(0, 0);
  int p = 0;
  for (int s = 0; s < NS; s++) {
    if (s + 1 < NS) {
      issue(s + 1, p ^ 1);   // overlap next-stage copy with this stage's WMMA
      WAIT_ASYNC(8);         // the 8 oldest (stage s) are complete
    } else {
      WAIT_ASYNC(0);
    }
    __syncthreads();

    v16h af[4], bf[4];
#pragma unroll
    for (int i = 0; i < 4; i++) {
      const _Float16* ap = &sA[p][mL + i * 16 + r][0];
      af[i] = frag_from2(ap + half * 8, ap + 16 + half * 8);
      const _Float16* bp = &sB[p][nL + i * 16 + r][0];
      bf[i] = frag_from2(bp + half * 8, bp + 16 + half * 8);
    }
#pragma unroll
    for (int i = 0; i < 4; i++)
#pragma unroll
      for (int j = 0; j < 4; j++)
        acc[i][j] = WMMA_F16(af[i], bf[j], acc[i][j]);

    __syncthreads();
    p ^= 1;
  }

  const int mW = mBlock + mL, nW = nBlock + nL;
#pragma unroll
  for (int i = 0; i < 4; i++) {
#pragma unroll
    for (int j = 0; j < 4; j++) {
      const int col = nW + j * 16 + r;
      const float bcol = bias[col];
#pragma unroll
      for (int v = 0; v < 8; v++) {
        const int row = mW + i * 16 + v + 8 * half;  // C/D layout: M = vgpr + 8*half
        const size_t idx = (size_t)row * N + col;
        float c = acc[i][j][v] + bcol;
        if constexpr (EPI == EPI_H16) {
          Ch[idx] = (_Float16)c;
        } else if constexpr (EPI == EPI_RES) {
          Cf[idx] = resid[idx] + c;
        } else {  // exact GELU
          float g = 0.5f * c * (1.0f + erff(c * 0.70710678118654752f));
          Ch[idx] = (_Float16)g;
        }
      }
    }
  }
}

// ---------------------------------------------------------------------------
// Repack V from qkv16[M,3C] into Vt[b][h][d][T] (f16) so P@V B-fragments
// are contiguous along the KV axis.
// ---------------------------------------------------------------------------
__global__ __launch_bounds__(256)
void k_repack_v(const _Float16* __restrict__ qkv, _Float16* __restrict__ Vt,
                int T, int H, long long total) {
  const int HD = 64;
  const int C = H * HD, C3 = 3 * C;
  long long idx = (long long)blockIdx.x * 256 + threadIdx.x;
  if (idx >= total) return;
  int t = (int)(idx % T);
  long long rest = idx / T;
  int d = (int)(rest % HD); rest /= HD;
  int h = (int)(rest % H);
  long long b = rest / H;
  Vt[idx] = qkv[(b * T + t) * (long long)C3 + 2 * C + h * HD + d];
}

// ---------------------------------------------------------------------------
// Causal flash attention, head_dim = 64, f16 in / f16 out, fp32 online softmax.
// Block = 128 threads = 4 waves; each wave owns 16 query rows.
// S = Q@K^T via WMMA; P transposed through per-wave LDS tile; O += P@V via WMMA.
// ---------------------------------------------------------------------------
__global__ __launch_bounds__(128)
void k_attn(const _Float16* __restrict__ qkv, const _Float16* __restrict__ Vt,
            _Float16* __restrict__ ctx, int T, int H) {
  const int HD = 64;
  const int C = H * HD, C3 = 3 * C;
  const int lane = threadIdx.x & 31;
  const int wave = threadIdx.x >> 5;
  const int r = lane & 15, half = lane >> 4;
  const int b = blockIdx.x / H, h = blockIdx.x % H;
  const int qBase = blockIdx.y * 64 + wave * 16;

  const _Float16* Qp  = qkv + (size_t)b * T * C3 + h * HD;
  const _Float16* Kp  = qkv + (size_t)b * T * C3 + C + h * HD;
  const _Float16* Vtp = Vt + ((size_t)(b * H + h) * HD) * T;

  __shared__ __align__(16) _Float16 ldsP[4][16][40];

  v16h aq[2];
  {
    const _Float16* qr = Qp + (size_t)(qBase + r) * C3;
#pragma unroll
    for (int s = 0; s < 2; s++)
      aq[s] = frag_from2(qr + s * 32 + half * 8, qr + s * 32 + 16 + half * 8);
  }

  v8f o[4] = {};
  float m[8], l[8];
#pragma unroll
  for (int v = 0; v < 8; v++) { m[v] = -INFINITY; l[v] = 0.0f; }
  const float scale = 0.125f;  // 1/sqrt(64)

  for (int kv = 0; kv < qBase + 16; kv += 32) {
    v8f s0 = {}, s1 = {};
#pragma unroll
    for (int s = 0; s < 2; s++) {
      const _Float16* k0 = Kp + (size_t)(kv +      r) * C3 + s * 32;
      const _Float16* k1 = Kp + (size_t)(kv + 16 + r) * C3 + s * 32;
      v16h b0 = frag_from2(k0 + half * 8, k0 + 16 + half * 8);
      v16h b1 = frag_from2(k1 + half * 8, k1 + 16 + half * 8);
      s0 = WMMA_F16(aq[s], b0, s0);
      s1 = WMMA_F16(aq[s], b1, s1);
    }
    float sv0[8], sv1[8];
#pragma unroll
    for (int v = 0; v < 8; v++) {
      const int row = qBase + v + 8 * half;
      sv0[v] = (kv +      r <= row) ? s0[v] * scale : -INFINITY;
      sv1[v] = (kv + 16 + r <= row) ? s1[v] * scale : -INFINITY;
    }
    float alpha[8];
#pragma unroll
    for (int v = 0; v < 8; v++) {
      float mx = fmaxf(sv0[v], sv1[v]);
#pragma unroll
      for (int off = 1; off < 16; off <<= 1) mx = fmaxf(mx, __shfl_xor(mx, off));
      const float mn = fmaxf(m[v], mx);
      alpha[v] = (m[v] == -INFINITY) ? 0.0f : __expf(m[v] - mn);
      m[v] = mn;
    }
#pragma unroll
    for (int v = 0; v < 8; v++) {
      const float p0 = (sv0[v] == -INFINITY) ? 0.0f : __expf(sv0[v] - m[v]);
      const float p1 = (sv1[v] == -INFINITY) ? 0.0f : __expf(sv1[v] - m[v]);
      float rs = p0 + p1;
#pragma unroll
      for (int off = 1; off < 16; off <<= 1) rs += __shfl_xor(rs, off);
      l[v] = l[v] * alpha[v] + rs;
#pragma unroll
      for (int d = 0; d < 4; d++) o[d][v] = o[d][v] * alpha[v];
      ldsP[wave][v + 8 * half][r]      = (_Float16)p0;
      ldsP[wave][v + 8 * half][16 + r] = (_Float16)p1;
    }
    __builtin_amdgcn_wave_barrier();
    asm volatile("s_wait_dscnt 0" ::: "memory");  // per-wave LDS RAW ordering
    const _Float16* pr = &ldsP[wave][r][0];
    v16h ap = frag_from2(pr + half * 8, pr + 16 + half * 8);
#pragma unroll
    for (int dt = 0; dt < 4; dt++) {
      const _Float16* vp = Vtp + (size_t)(dt * 16 + r) * T + kv;
      v16h bv = frag_from2(vp + half * 8, vp + 16 + half * 8);
      o[dt] = WMMA_F16(ap, bv, o[dt]);
    }
    __builtin_amdgcn_wave_barrier();
  }

#pragma unroll
  for (int dt = 0; dt < 4; dt++) {
#pragma unroll
    for (int v = 0; v < 8; v++) {
      const int row = qBase + v + 8 * half;
      const float val = o[dt][v] / l[v];
      ctx[(size_t)(b * T + row) * C + h * HD + dt * 16 + r] = (_Float16)val;
    }
  }
}

// ---------------------------------------------------------------------------
// Driver
// ---------------------------------------------------------------------------
extern "C" void kernel_launch(void* const* d_in, const int* in_sizes, int n_in,
                              void* d_out, int out_size, void* d_ws, size_t ws_size,
                              hipStream_t stream) {
  (void)in_sizes; (void)n_in; (void)out_size; (void)ws_size;
  const int Bb = 4, T = 2048, H = 16, HD = 64;
  const int C = H * HD;        // 1024
  const int C3 = 3 * C, C4 = 4 * C;
  const int M = Bb * T;        // 8192

  const float* x      = (const float*)d_in[0];
  const float* ln1_w  = (const float*)d_in[1];
  const float* ln1_b  = (const float*)d_in[2];
  const float* w_qkv  = (const float*)d_in[3];
  const float* b_qkv  = (const float*)d_in[4];
  const float* w_o    = (const float*)d_in[5];
  const float* b_o    = (const float*)d_in[6];
  const float* ln2_w  = (const float*)d_in[7];
  const float* ln2_b  = (const float*)d_in[8];
  const float* w_fc   = (const float*)d_in[9];
  const float* b_fc   = (const float*)d_in[10];
  const float* w_proj = (const float*)d_in[11];
  const float* b_proj = (const float*)d_in[12];
  float* out = (float*)d_out;

  char* ws = (char*)d_ws;
  size_t off = 0;
  auto alloc = [&](size_t bytes) -> void* {
    void* p = ws + off;
    off = (off + bytes + 255) & ~(size_t)255;
    return p;
  };
  _Float16* wqkvT  = (_Float16*)alloc((size_t)C3 * C  * sizeof(_Float16));
  _Float16* woT    = (_Float16*)alloc((size_t)C  * C  * sizeof(_Float16));
  _Float16* wfcT   = (_Float16*)alloc((size_t)C4 * C  * sizeof(_Float16));
  _Float16* wprojT = (_Float16*)alloc((size_t)C  * C4 * sizeof(_Float16));
  _Float16* xln    = (_Float16*)alloc((size_t)M * C  * sizeof(_Float16));
  _Float16* qkv16  = (_Float16*)alloc((size_t)M * C3 * sizeof(_Float16));
  _Float16* vT     = (_Float16*)alloc((size_t)M * C  * sizeof(_Float16));
  _Float16* ctx16  = (_Float16*)alloc((size_t)M * C  * sizeof(_Float16));
  _Float16* h16    = (_Float16*)alloc((size_t)M * C4 * sizeof(_Float16));
  float*    x1     = (float*)   alloc((size_t)M * C  * sizeof(float));

  // 1) weight cast + transpose (f32 [K,N] -> f16 [N,K]); all fit in 192MB L2
  k_transpose_cast<<<dim3(C3 / 32, C / 32),  256, 0, stream>>>(w_qkv,  wqkvT,  C,  C3);
  k_transpose_cast<<<dim3(C  / 32, C / 32),  256, 0, stream>>>(w_o,    woT,    C,  C);
  k_transpose_cast<<<dim3(C4 / 32, C / 32),  256, 0, stream>>>(w_fc,   wfcT,   C,  C4);
  k_transpose_cast<<<dim3(C  / 32, C4 / 32), 256, 0, stream>>>(w_proj, wprojT, C4, C);

  // 2) ln1 -> f16
  k_ln_cast<<<M, 256, 0, stream>>>(x, ln1_w, ln1_b, xln, C);
  // 3) qkv = ln1(x) @ w_qkv + b_qkv   (f16 out)
  k_gemm<EPI_H16><<<dim3(C3 / 128, M / 128), 128, 0, stream>>>(
      xln, wqkvT, b_qkv, nullptr, nullptr, qkv16, M, C3, C);
  // 4) repack V -> [b,h,d,T]
  {
    long long total = (long long)M * C;
    k_repack_v<<<(unsigned)((total + 255) / 256), 256, 0, stream>>>(qkv16, vT, T, H, total);
  }
  // 5) causal flash attention -> ctx f16
  k_attn<<<dim3(Bb * H, T / 64), 128, 0, stream>>>(qkv16, vT, ctx16, T, H);
  // 6) x1 = x + ctx @ w_o + b_o   (fp32)
  k_gemm<EPI_RES><<<dim3(C / 128, M / 128), 128, 0, stream>>>(
      ctx16, woT, b_o, x, x1, nullptr, M, C, C);
  // 7) ln2 -> f16
  k_ln_cast<<<M, 256, 0, stream>>>(x1, ln2_w, ln2_b, xln, C);
  // 8) h = gelu(ln2 @ w_fc + b_fc)   (f16 out)
  k_gemm<EPI_GELU><<<dim3(C4 / 128, M / 128), 128, 0, stream>>>(
      xln, wfcT, b_fc, nullptr, nullptr, h16, M, C4, C);
  // 9) out = x1 + h @ w_proj + b_proj   (fp32 -> d_out)
  k_gemm<EPI_RES><<<dim3(C / 128, M / 128), 128, 0, stream>>>(
      h16, wprojT, b_proj, x1, out, nullptr, M, C, C4);
}
